// LinearAttentionWithKQV_28063316312365
// MI455X (gfx1250) — matrix operand order repaired and checked by reading, not compiled, not run on hardware
//
#include <hip/hip_runtime.h>

typedef __bf16 bf16;
typedef __attribute__((ext_vector_type(16))) __bf16 v16bf;
typedef __attribute__((ext_vector_type(8)))  __bf16 v8bf;
typedef __attribute__((ext_vector_type(4)))  __bf16 v4bf;
typedef __attribute__((ext_vector_type(8)))  float   v8f;
typedef __attribute__((ext_vector_type(4)))  float   f32x4;
typedef unsigned int u32;
typedef __attribute__((ext_vector_type(4))) u32 u32x4;
typedef __attribute__((ext_vector_type(8))) int i32x8;
typedef __attribute__((ext_vector_type(4))) int i32x4;

#define BDIM 8
#define SDIM 4096
#define DDIM 1024

#ifndef __has_builtin
#define __has_builtin(x) 0
#endif
#if __has_builtin(__builtin_amdgcn_tensor_load_to_lds) && __has_builtin(__builtin_amdgcn_s_wait_tensorcnt)
#define USE_TDM 1
#else
#define USE_TDM 0
#endif

// ---------------------------------------------------------------------------
// WMMA fragment loads from LDS (wave32, v_wmma_f32_16x16x32_bf16 layouts).
// LDS tiles are [128 rows x 64 k] bf16 with row pitch 72 elems (144B):
// 16B-aligned fragment loads, lane stride 36 dwords -> all 64 banks covered.
// ---------------------------------------------------------------------------
__device__ __forceinline__ v16bf frag_a(const bf16* s, int row, int kb, int lane) {
    const int hf = lane >> 4, lr = lane & 15;
    const bf16* p = s + (row + lr) * 72 + kb + hf * 8;
    v8bf lo = *(const v8bf*)p;
    v8bf hi = *(const v8bf*)(p + 16);
    v16bf f;
#pragma unroll
    for (int i = 0; i < 8; ++i) { f[i] = lo[i]; f[i + 8] = hi[i]; }
    return f;
}

__device__ __forceinline__ v16bf frag_b(const bf16* s, int row, int kb, int lane) {
    const int hf = lane >> 4, lr = lane & 15;
    const bf16* p = s + (row + lr) * 72 + kb + hf * 8;
    v8bf lo = *(const v8bf*)p;
    v8bf hi = *(const v8bf*)(p + 16);
    v16bf f;
#pragma unroll
    for (int i = 0; i < 8; ++i) { f[i] = lo[i]; f[i + 8] = hi[i]; }
    return f;
}

#if USE_TDM
// ---------------------------------------------------------------------------
// Tensor Data Mover: DMA one [128 x 64] bf16 tile (row stride ld elems) from
// global into LDS at lds_off, inserting 4 dwords of pad after every 32 dwords
// (one 128B row) -> LDS row pitch 72 elems. D# per CDNA5 ISA 8.3-8.6.
// This toolchain exposes the 6-arg builtin (extra i32x8 group, zero-filled).
// ---------------------------------------------------------------------------
__device__ __forceinline__ void tdm_load_tile(const bf16* gsrc, u32 lds_off, int ld) {
    const unsigned long long ga = (unsigned long long)(size_t)gsrc;
    u32x4 g0;
    g0[0] = 1u;                                     // count=1, user mode, no gather
    g0[1] = lds_off;                                // lds_addr (bytes)
    g0[2] = (u32)(ga & 0xffffffffu);                // global_addr[31:0]
    g0[3] = (u32)((ga >> 32) & 0x1ffffffu) | (2u << 30);  // global_addr[56:32] | type=2

    const u32 td = 0x7fffffffu;                     // huge tensor dims: no OOB clip
    i32x8 g1;
    g1[0] = (int)((1u << 16) |                      // data_size = 2 bytes
                  (1u << 20) |                      // pad_enable
                  (4u << 22) |                      // pad_interval: every 32 dwords
                  (3u << 25));                      // pad_amount: 4 dwords (8 elems)
    g1[1] = (int)((td & 0xffffu) << 16);            // tensor_dim0[15:0]
    g1[2] = (int)((td >> 16) | ((td & 0xffffu) << 16));   // td0[31:16] | td1[15:0]
    g1[3] = (int)((td >> 16) | (64u << 16));        // td1[31:16] | tile_dim0=64
    g1[4] = (int)128u;                              // tile_dim1=128, tile_dim2=0
    g1[5] = (int)(u32)ld;                           // tensor_dim0_stride[31:0] (elems)
    g1[6] = 0;                                      // stride0[47:32] | stride1[15:0]
    g1[7] = 0;
    i32x4 g2; g2[0] = 1; g2[1] = 0; g2[2] = 0; g2[3] = 0;  // tensor_dim2=1
    i32x4 g3; g3[0] = 0; g3[1] = 0; g3[2] = 0; g3[3] = 0;
    i32x8 g4;                                        // extra group (clang-23 form)
#pragma unroll
    for (int i = 0; i < 8; ++i) g4[i] = 0;
    __builtin_amdgcn_tensor_load_to_lds(g0, g1, g2, g3, g4, 0);
}
#endif

// fallback / non-TDM tile fill: 256 threads copy [128 x 64] bf16 -> pitch 72
__device__ __forceinline__ void fill_tile(bf16* s, const bf16* G, int ld, int tid) {
#pragma unroll
    for (int it = 0; it < 4; ++it) {
        int idx = tid + it * 256;
        int row = idx >> 3, c8 = idx & 7;
        v8bf x = *(const v8bf*)(G + (long)row * ld + c8 * 8);
        *(v8bf*)(s + row * 72 + c8 * 8) = x;
    }
}

// ---------------------------------------------------------------------------
// C = scale * (A @ B^T) [+ bias]: A [M x K] bf16 (ld=lda), B^T rows [N x K]
// bf16 (ld=ldb). 128x128 tile / WG, BLOCK_K=64, 8 waves of 64x32 each.
// OUTMODE: 0 = f32 row-major, 1 = bf16 row-major, 2 = bf16 transposed C^T[n][m]
// (mode 2 uses one 16B store per accumulator: lane's 8 VGPRs = 8 consecutive M)
// ---------------------------------------------------------------------------
template<bool HAS_BIAS, int OUTMODE>
__global__ __launch_bounds__(256) void gemm_bf16(
    const bf16* __restrict__ Ag, const bf16* __restrict__ Bg,
    const float* __restrict__ bias, void* __restrict__ Cout,
    int K, int lda, int ldb, int ldc, float scale,
    long strideA, long strideB, long strideC)
{
    __shared__ bf16 sA[2][128 * 72];
    __shared__ bf16 sB[2][128 * 72];

    const int tid = threadIdx.x;
    const int lane = tid & 31, wid = tid >> 5;
    const int wm = (wid >> 2) * 64, wn = (wid & 3) * 32;
    const int m0 = blockIdx.y * 128, n0 = blockIdx.x * 128;
    const int bz = blockIdx.z;
    const bf16* Abase = Ag + (long)bz * strideA + (long)m0 * lda;
    const bf16* Bbase = Bg + (long)bz * strideB + (long)n0 * ldb;

    v8f acc[4][2];
#pragma unroll
    for (int i = 0; i < 4; ++i)
#pragma unroll
        for (int j = 0; j < 2; ++j)
#pragma unroll
            for (int e = 0; e < 8; ++e) acc[i][j][e] = 0.0f;

    const int KT = K >> 6;

#if USE_TDM
    if (wid == 0) {
        tdm_load_tile(Abase, (u32)(size_t)&sA[0][0], lda);
        tdm_load_tile(Bbase, (u32)(size_t)&sB[0][0], ldb);
    }
#endif

    for (int kt = 0; kt < KT; ++kt) {
#if USE_TDM
        const int cur = kt & 1;
        __syncthreads();                       // prev reads of buf[1-cur] done
        if (wid == 0) {
            if (kt + 1 < KT) {
                tdm_load_tile(Abase + (kt + 1) * 64, (u32)(size_t)&sA[1 - cur][0], lda);
                tdm_load_tile(Bbase + (kt + 1) * 64, (u32)(size_t)&sB[1 - cur][0], ldb);
                __builtin_amdgcn_s_wait_tensorcnt(2);   // buf[cur] tiles retired
            } else {
                __builtin_amdgcn_s_wait_tensorcnt(0);
            }
        }
        __syncthreads();                       // buf[cur] ready for all waves
#else
        const int cur = 0;
        __syncthreads();
        fill_tile(sA[0], Abase + kt * 64, lda, tid);
        fill_tile(sB[0], Bbase + kt * 64, ldb, tid);
        __syncthreads();
#endif
        const bf16* sAc = &sA[cur][0];
        const bf16* sBc = &sB[cur][0];
#pragma unroll
        for (int ks = 0; ks < 2; ++ks) {
            const int kb = ks * 32;
            v16bf af[4], bfr[2];
#pragma unroll
            for (int mi = 0; mi < 4; ++mi) af[mi] = frag_a(sAc, wm + mi * 16, kb, lane);
#pragma unroll
            for (int ni = 0; ni < 2; ++ni) bfr[ni] = frag_b(sBc, wn + ni * 16, kb, lane);
#pragma unroll
            for (int mi = 0; mi < 4; ++mi)
#pragma unroll
                for (int ni = 0; ni < 2; ++ni)
                    acc[mi][ni] = __builtin_amdgcn_wmma_f32_16x16x32_bf16(
                        false, af[mi], false, bfr[ni], (short)0, acc[mi][ni], false, false);
        }
    }

    const int hf = lane >> 4, cn = lane & 15;
#pragma unroll
    for (int mi = 0; mi < 4; ++mi) {
#pragma unroll
        for (int ni = 0; ni < 2; ++ni) {
            const int gm = m0 + wm + mi * 16 + hf * 8;
            const int gn = n0 + wn + ni * 16 + cn;
            const float bv = HAS_BIAS ? bias[gn] : 0.0f;
            if constexpr (OUTMODE == 2) {
                v8bf o;
#pragma unroll
                for (int r = 0; r < 8; ++r) o[r] = (bf16)(acc[mi][ni][r] * scale + bv);
                *(v8bf*)((bf16*)Cout + (long)bz * strideC + (long)gn * ldc + gm) = o;
            } else if constexpr (OUTMODE == 1) {
#pragma unroll
                for (int r = 0; r < 8; ++r)
                    ((bf16*)Cout)[(long)bz * strideC + (long)(gm + r) * ldc + gn] =
                        (bf16)(acc[mi][ni][r] * scale + bv);
            } else {
#pragma unroll
                for (int r = 0; r < 8; ++r)
                    ((float*)Cout)[(long)bz * strideC + (long)(gm + r) * ldc + gn] =
                        acc[mi][ni][r] * scale + bv;
            }
        }
    }
}

// ---------------------------------------------------------------------------
__global__ __launch_bounds__(256) void cvt_f32_bf16(
    const float* __restrict__ in, bf16* __restrict__ out, long n)
{
    const long i = ((long)blockIdx.x * 256 + threadIdx.x) * 4;
    if (i < n) {
        f32x4 x = *(const f32x4*)(in + i);
        v4bf o;
        o[0] = (bf16)x[0]; o[1] = (bf16)x[1]; o[2] = (bf16)x[2]; o[3] = (bf16)x[3];
        *(v4bf*)(out + i) = o;
    }
}

// ---------------------------------------------------------------------------
// In-place softmax of Q over the sequence axis: per (b, d) column of length S.
// ---------------------------------------------------------------------------
__global__ __launch_bounds__(256) void softmax_seq(bf16* __restrict__ Q)
{
    __shared__ float red[4][64];
    const int tid = threadIdx.x;
    const int dOff = tid & 63, sl = tid >> 6;
    const int b = blockIdx.x >> 4;            // DDIM/64 = 16 WGs per batch
    const int dBase = (blockIdx.x & 15) * 64;
    bf16* col = Q + (long)b * SDIM * DDIM + dBase + dOff;

    float m = -3.0e38f;
    for (int s = sl; s < SDIM; s += 4) m = fmaxf(m, (float)col[(long)s * DDIM]);
    red[sl][dOff] = m;
    __syncthreads();
    m = fmaxf(fmaxf(red[0][dOff], red[1][dOff]), fmaxf(red[2][dOff], red[3][dOff]));
    __syncthreads();

    float sum = 0.0f;
    for (int s = sl; s < SDIM; s += 4) sum += __expf((float)col[(long)s * DDIM] - m);
    red[sl][dOff] = sum;
    __syncthreads();
    sum = red[0][dOff] + red[1][dOff] + red[2][dOff] + red[3][dOff];
    const float inv = 1.0f / sum;

    for (int s = sl; s < SDIM; s += 4) {
        const long off = (long)s * DDIM;
        col[off] = (bf16)(__expf((float)col[off] - m) * inv);
    }
}

// ---------------------------------------------------------------------------
extern "C" void kernel_launch(void* const* d_in, const int* in_sizes, int n_in,
                              void* d_out, int out_size, void* d_ws, size_t ws_size,
                              hipStream_t stream)
{
    (void)in_sizes; (void)n_in; (void)out_size; (void)ws_size;
    const float* X  = (const float*)d_in[0];
    const float* Wk = (const float*)d_in[1];
    const float* bk = (const float*)d_in[2];
    const float* Wq = (const float*)d_in[3];
    const float* bq = (const float*)d_in[4];
    const float* Wv = (const float*)d_in[5];
    const float* bv = (const float*)d_in[6];

    const long BSD = (long)BDIM * SDIM * DDIM;      // 33,554,432
    const long DD  = (long)DDIM * DDIM;             // 1,048,576
    bf16* Xb  = (bf16*)d_ws;                        // 64 MB
    bf16* Wkb = Xb  + BSD;                          // 2 MB
    bf16* Wqb = Wkb + DD;
    bf16* Wvb = Wqb + DD;
    bf16* KT  = Wvb + DD;                           // K^T [B][D][S], 64 MB
    bf16* VT  = KT  + BSD;                          // V^T [B][D][S], 64 MB
    bf16* Qb  = VT  + BSD;                          // Q   [B][S][D], 64 MB
    bf16* Sc  = Qb  + BSD;                          // scores^T [B][e][d], 16 MB

    const dim3 blk(256);
    const dim3 gProj(DDIM / 128, SDIM / 128, BDIM); // (8, 32, 8)
    const dim3 gScore(DDIM / 128, DDIM / 128, BDIM);// (8, 8, 8)
    const long sIn = (long)SDIM * DDIM;
    const float invRootD = 0.03125f;                // 1/sqrt(1024)

    // f32 -> bf16 once (halves X traffic for the 3 projections; TDM-friendly)
    cvt_f32_bf16<<<dim3((unsigned)(BSD / 1024)), blk, 0, stream>>>(X, Xb, BSD);
    cvt_f32_bf16<<<dim3((unsigned)(DD / 1024)), blk, 0, stream>>>(Wk, Wkb, DD);
    cvt_f32_bf16<<<dim3((unsigned)(DD / 1024)), blk, 0, stream>>>(Wq, Wqb, DD);
    cvt_f32_bf16<<<dim3((unsigned)(DD / 1024)), blk, 0, stream>>>(Wv, Wvb, DD);

    // K^T and V^T via transposed epilogue (16B stores); Q row-major for softmax
    gemm_bf16<true, 2><<<gProj, blk, 0, stream>>>(
        Xb, Wkb, bk, KT, DDIM, DDIM, DDIM, SDIM, 1.0f, sIn, 0L, sIn);
    gemm_bf16<true, 2><<<gProj, blk, 0, stream>>>(
        Xb, Wvb, bv, VT, DDIM, DDIM, DDIM, SDIM, 1.0f, sIn, 0L, sIn);
    gemm_bf16<true, 1><<<gProj, blk, 0, stream>>>(
        Xb, Wqb, bq, Qb, DDIM, DDIM, DDIM, DDIM, 1.0f, sIn, 0L, sIn);

    // softmax(Q) over sequence axis, in place
    softmax_seq<<<dim3(BDIM * (DDIM / 64)), blk, 0, stream>>>(Qb);

    // C[d][e] = sum_s K^T[d][s] V^T[e][s] / sqrt(D); transposed store -> Sc[e][d]
    gemm_bf16<false, 2><<<gScore, blk, 0, stream>>>(
        KT, VT, nullptr, Sc, SDIM, SDIM, SDIM, DDIM, invRootD, sIn, sIn, DD);

    // out[s][e] = sum_d softQ[s][d] * Sc[e][d] / sqrt(D), f32 row-major
    gemm_bf16<false, 0><<<gProj, blk, 0, stream>>>(
        Qb, Sc, nullptr, d_out, DDIM, DDIM, DDIM, DDIM, invRootD, sIn, DD, sIn);
}
